// MSA_Col_Att_34660386079045
// MI455X (gfx1250) — matrix operand order repaired and checked by reading, not compiled, not run on hardware
//
// MSA column attention for MI455X (gfx1250), wave32 + WMMA bf16.
//
// Pipeline: k_ln (LN->bf16 X) -> k_wprep (weightsT->bf16) ->
//           k_proj (X @ [Wq|Wk|Wv|Wg], WMMA, writes [L][H][S][Dh] slabs) ->
//           k_att  (per (c,n) flash attention, WMMA QK^T and PV) ->
//           k_out  (O @ Wo^T + b_o, WMMA, f32 out).
// ~60 GFLOP total, ~300MB HBM traffic => balanced; bf16 WMMA everywhere.

#include <hip/hip_runtime.h>

#define S_DIM 256
#define L_DIM 256
#define D_DIM 256
#define H_NUM 8
#define DH 32
#define INNER 256
#define NROWS 65536           // S*L

typedef __attribute__((ext_vector_type(16))) __bf16 v16bf;
typedef __attribute__((ext_vector_type(8)))  float  v8f;

struct __align__(16) U4 { unsigned int x, y, z, w; };
struct __align__(16) F4 { float x, y, z, w; };

union Frag {
    v16bf bf;
    U4 q[2];
    unsigned int u[8];
};

__device__ __forceinline__ unsigned short f2bf(float f) {
    union { float f; unsigned int u; } a; a.f = f;
    unsigned int u = a.u;
    u += 0x7FFFu + ((u >> 16) & 1u);   // RNE
    return (unsigned short)(u >> 16);
}
__device__ __forceinline__ float bf2f(unsigned short h) {
    union { unsigned int u; float f; } a; a.u = ((unsigned int)h) << 16;
    return a.f;
}

// ---------------------------------------------------------------------------
// Kernel 1: LayerNorm  (one row of 256 f32 per wave) -> bf16 Xln
// ---------------------------------------------------------------------------
__global__ __launch_bounds__(256) void k_ln(const float* __restrict__ msa,
                                            const float* __restrict__ gamma,
                                            const float* __restrict__ beta,
                                            unsigned short* __restrict__ xln) {
    const int t = threadIdx.x, lane = t & 31, w = t >> 5;
    const int row = blockIdx.x * 8 + w;
    const float* rp = msa + (size_t)row * D_DIM + lane * 8;
    F4 x0 = *(const F4*)rp;
    F4 x1 = *(const F4*)(rp + 4);
    float s  = x0.x + x0.y + x0.z + x0.w + x1.x + x1.y + x1.z + x1.w;
    float s2 = x0.x*x0.x + x0.y*x0.y + x0.z*x0.z + x0.w*x0.w
             + x1.x*x1.x + x1.y*x1.y + x1.z*x1.z + x1.w*x1.w;
    #pragma unroll
    for (int off = 16; off >= 1; off >>= 1) {
        s  += __shfl_xor(s,  off, 32);
        s2 += __shfl_xor(s2, off, 32);
    }
    const float mean = s * (1.0f / 256.0f);
    const float var  = s2 * (1.0f / 256.0f) - mean * mean;
    const float rstd = rsqrtf(var + 1e-5f);
    F4 g0 = *(const F4*)(gamma + lane * 8);
    F4 g1 = *(const F4*)(gamma + lane * 8 + 4);
    F4 b0 = *(const F4*)(beta  + lane * 8);
    F4 b1 = *(const F4*)(beta  + lane * 8 + 4);
    float y[8];
    y[0]=(x0.x-mean)*rstd*g0.x+b0.x; y[1]=(x0.y-mean)*rstd*g0.y+b0.y;
    y[2]=(x0.z-mean)*rstd*g0.z+b0.z; y[3]=(x0.w-mean)*rstd*g0.w+b0.w;
    y[4]=(x1.x-mean)*rstd*g1.x+b1.x; y[5]=(x1.y-mean)*rstd*g1.y+b1.y;
    y[6]=(x1.z-mean)*rstd*g1.z+b1.z; y[7]=(x1.w-mean)*rstd*g1.w+b1.w;
    U4 o;
    o.x = (unsigned)f2bf(y[0]) | ((unsigned)f2bf(y[1]) << 16);
    o.y = (unsigned)f2bf(y[2]) | ((unsigned)f2bf(y[3]) << 16);
    o.z = (unsigned)f2bf(y[4]) | ((unsigned)f2bf(y[5]) << 16);
    o.w = (unsigned)f2bf(y[6]) | ((unsigned)f2bf(y[7]) << 16);
    *(U4*)(xln + (size_t)row * D_DIM + lane * 8) = o;
}

// ---------------------------------------------------------------------------
// Kernel 2: weight transpose + f32->bf16.  out[n][k] = in[k][n]
// p = 0..3 -> Wq,Wk,Wv,Wg into wt[p];  p = 4 -> Wo into wot
// ---------------------------------------------------------------------------
__global__ __launch_bounds__(256) void k_wprep(const float* __restrict__ wq,
                                               const float* __restrict__ wk,
                                               const float* __restrict__ wv,
                                               const float* __restrict__ wg,
                                               const float* __restrict__ wo,
                                               unsigned short* __restrict__ wt,
                                               unsigned short* __restrict__ wot) {
    __shared__ float lds[32][33];
    const int p = blockIdx.y;
    const int bt = blockIdx.x;
    const int tk = (bt >> 3) * 32, tn = (bt & 7) * 32;
    const int t = threadIdx.x, r0 = t >> 5, cc = t & 31;
    const float* src = (p == 0) ? wq : (p == 1) ? wk : (p == 2) ? wv
                     : (p == 3) ? wg : wo;
    unsigned short* dst = (p < 4) ? (wt + p * 65536) : wot;
    #pragma unroll
    for (int i = 0; i < 4; ++i) {
        int r = r0 + i * 8;
        lds[r][cc] = src[(size_t)(tk + r) * 256 + tn + cc];
    }
    __syncthreads();
    #pragma unroll
    for (int i = 0; i < 4; ++i) {
        int r = r0 + i * 8;
        dst[(size_t)(tn + r) * 256 + tk + cc] = f2bf(lds[cc][r]);
    }
}

// ---------------------------------------------------------------------------
// Kernel 3: projection GEMM. block = 64 rows x 256 cols of projection p.
// A from LDS (row stride 528B, conflict-free b128 fragment reads),
// B from global transposed weights (L2 resident).
// Output scattered to [L][H][S][Dh] bf16 slabs (attention-friendly).
// ---------------------------------------------------------------------------
__global__ __launch_bounds__(256) void k_proj(const unsigned short* __restrict__ xln,
                                              const unsigned short* __restrict__ wt,
                                              const float* __restrict__ b_g,
                                              unsigned short* __restrict__ qkvg) {
    __shared__ __align__(16) unsigned char smem[64 * 528];
    const int rb = blockIdx.x * 64;
    const int p  = blockIdx.y;
    const int t = threadIdx.x, lane = t & 31, w = t >> 5;
    // Stage 64 rows x 256 bf16 of Xln into LDS (pad each 512B row to 528B).
    #pragma unroll
    for (int i = 0; i < 8; ++i) {
        int idx = t + i * 256;           // 2048 x 16B chunks
        int row = idx >> 5, qq = idx & 31;
        U4 d = *(const U4*)(xln + (size_t)(rb + row) * 256 + qq * 8);
        *(U4*)(smem + row * 528 + qq * 16) = d;
    }
    __syncthreads();
    const int l15 = lane & 15, hiw = lane >> 4;
    const int mt = w & 3, nh = w >> 2;       // wave: 1 Mtile x 8 Ntiles
    v8f acc[8];
    v8f z = {0.f,0.f,0.f,0.f,0.f,0.f,0.f,0.f};
    #pragma unroll
    for (int i = 0; i < 8; ++i) acc[i] = z;
    const unsigned short* wp = wt + p * 65536;
    #pragma unroll
    for (int ks = 0; ks < 8; ++ks) {
        Frag a;  // A frag: lane=M row; K chunks (hi?8:0)+0..7 and +16..23
        const unsigned char* ap = smem + (mt * 16 + l15) * 528 + ks * 64 + hiw * 16;
        a.q[0] = *(const U4*)ap;
        a.q[1] = *(const U4*)(ap + 32);
        #pragma unroll
        for (int nt = 0; nt < 8; ++nt) {
            int ncol = (nh * 8 + nt) * 16 + l15;  // B frag: lane=N col
            const unsigned short* bp = wp + (size_t)ncol * 256 + ks * 32 + hiw * 8;
            Frag b;
            b.q[0] = *(const U4*)bp;
            b.q[1] = *(const U4*)(bp + 16);
            acc[nt] = __builtin_amdgcn_wmma_f32_16x16x32_bf16(
                false, a.bf, false, b.bf, (short)0, acc[nt], false, false);
        }
    }
    // Epilogue: C layout (VGPR v = row M (+8 upper half), lane = N col).
    #pragma unroll
    for (int nt = 0; nt < 8; ++nt) {
        int j = (nh * 8 + nt) * 16 + l15;
        int nhd = j >> 5, hh = j & 31;
        float bg = (p == 3) ? b_g[j] : 0.0f;
        #pragma unroll
        for (int v = 0; v < 8; ++v) {
            int r = rb + mt * 16 + v + hiw * 8;
            int s = r >> 8, c = r & 255;
            float val = acc[nt][v];
            if (p == 0) val *= 0.17677669529663687f;            // 1/sqrt(32)
            if (p == 3) val = 1.0f / (1.0f + __expf(-(val + bg)));
            size_t idx = (((size_t)p) << 24) +
                         ((size_t)((c * 8 + nhd) * 256 + s)) * 32 + hh;
            qkvg[idx] = f2bf(val);
        }
    }
}

// ---------------------------------------------------------------------------
// Kernel 4: attention, one block per (column c, head n). 8 waves x 32 rows.
// LDS: K rows padded 80B (K^T B-frags), V transposed [h][j] stride 528B
// (PV B-frags contiguous), per-wave P scratch (C-layout -> A-layout bridge).
// Online softmax in C-fragment layout, stats reduced over 16-lane halves.
// ---------------------------------------------------------------------------
#define ALD_K  0
#define ALD_VT 20480            // 256*80
#define ALD_P  37376            // + 32*528
__global__ __launch_bounds__(256) void k_att(const unsigned short* __restrict__ qkvg,
                                             unsigned short* __restrict__ obuf) {
    __shared__ __align__(16) unsigned char smem[57856];
    const int cn = blockIdx.x;
    const int c = cn >> 3, n = cn & 7;
    const int t = threadIdx.x, lane = t & 31, w = t >> 5;
    const unsigned short* Qt = qkvg + (size_t)cn * 8192;
    const unsigned short* Kt = qkvg + (1u << 24) + (size_t)cn * 8192;
    const unsigned short* Vt = qkvg + (2u << 24) + (size_t)cn * 8192;
    const unsigned short* Gt = qkvg + (3u << 24) + (size_t)cn * 8192;
    // Phase 1: stage K (row-major, 80B rows) and V transposed ([h][j]).
    #pragma unroll
    for (int i = 0; i < 4; ++i) {
        int idx = t + i * 256;              // 1024 x 16B chunks, 4 per 64B row
        int row = idx >> 2, qq = idx & 3;
        U4 kv = *(const U4*)(Kt + row * 32 + qq * 8);
        *(U4*)(smem + ALD_K + row * 80 + qq * 16) = kv;
        U4 vv = *(const U4*)(Vt + row * 32 + qq * 8);
        unsigned int uu[4] = { vv.x, vv.y, vv.z, vv.w };
        #pragma unroll
        for (int e = 0; e < 4; ++e) {
            int h0 = qq * 8 + e * 2;
            *(unsigned short*)(smem + ALD_VT + (h0 + 0) * 528 + row * 2) =
                (unsigned short)(uu[e] & 0xffffu);
            *(unsigned short*)(smem + ALD_VT + (h0 + 1) * 528 + row * 2) =
                (unsigned short)(uu[e] >> 16);
        }
    }
    __syncthreads();
    const int l15 = lane & 15, hiw = lane >> 4;
    unsigned char* ps = smem + ALD_P + w * 2560;   // 32 x 80B per wave
    float mstat[2][8], lstat[2][8];
    v8f oacc[2][2];
    v8f z = {0.f,0.f,0.f,0.f,0.f,0.f,0.f,0.f};
    #pragma unroll
    for (int mt = 0; mt < 2; ++mt) {
        oacc[mt][0] = z; oacc[mt][1] = z;
        #pragma unroll
        for (int v = 0; v < 8; ++v) { mstat[mt][v] = -1e30f; lstat[mt][v] = 0.f; }
    }
    // Q A-fragments for this wave's 32 rows (reused across all j-blocks).
    Frag qf[2];
    #pragma unroll
    for (int mt = 0; mt < 2; ++mt) {
        int s = w * 32 + mt * 16 + l15;
        const unsigned short* qp = Qt + s * 32 + hiw * 16;
        qf[mt].q[0] = *(const U4*)qp;
        qf[mt].q[1] = *(const U4*)(qp + 8);
    }
    for (int jb = 0; jb < 8; ++jb) {       // 32 keys per step
        Frag kf[2];
        #pragma unroll
        for (int nt = 0; nt < 2; ++nt) {   // B = K^T: lane=N(j), K=h contiguous
            int j = jb * 32 + nt * 16 + l15;
            const unsigned char* kp = smem + ALD_K + j * 80 + hiw * 32;
            kf[nt].q[0] = *(const U4*)kp;
            kf[nt].q[1] = *(const U4*)(kp + 16);
        }
        v8f sacc[2][2];
        #pragma unroll
        for (int mt = 0; mt < 2; ++mt)
            #pragma unroll
            for (int nt = 0; nt < 2; ++nt)
                sacc[mt][nt] = __builtin_amdgcn_wmma_f32_16x16x32_bf16(
                    false, qf[mt].bf, false, kf[nt].bf, (short)0, z, false, false);
        // Online softmax (row = v + hiw*8 within each Mtile; stats per half).
        #pragma unroll
        for (int mt = 0; mt < 2; ++mt) {
            #pragma unroll
            for (int v = 0; v < 8; ++v) {
                float rmax = fmaxf(sacc[mt][0][v], sacc[mt][1][v]);
                #pragma unroll
                for (int off = 8; off >= 1; off >>= 1)
                    rmax = fmaxf(rmax, __shfl_xor(rmax, off, 32));
                float mold = mstat[mt][v];
                float mnew = fmaxf(mold, rmax);
                float scale = __expf(mold - mnew);
                float p0 = __expf(sacc[mt][0][v] - mnew);
                float p1 = __expf(sacc[mt][1][v] - mnew);
                float rsum = p0 + p1;
                #pragma unroll
                for (int off = 8; off >= 1; off >>= 1)
                    rsum += __shfl_xor(rsum, off, 32);
                mstat[mt][v] = mnew;
                lstat[mt][v] = lstat[mt][v] * scale + rsum;
                oacc[mt][0][v] *= scale;
                oacc[mt][1][v] *= scale;
                int prow = mt * 16 + v + hiw * 8;  // C layout -> LDS scratch
                *(unsigned short*)(ps + prow * 80 + l15 * 2)        = f2bf(p0);
                *(unsigned short*)(ps + prow * 80 + (16 + l15) * 2) = f2bf(p1);
            }
        }
        // O += P @ V  (same-wave LDS RAW: DS ops in-order per wave).
        #pragma unroll
        for (int mt = 0; mt < 2; ++mt) {
            Frag pf;  // A frag from P scratch: lane=M row, K=j contiguous
            const unsigned char* pp = ps + (mt * 16 + l15) * 80 + hiw * 32;
            pf.q[0] = *(const U4*)pp;
            pf.q[1] = *(const U4*)(pp + 16);
            #pragma unroll
            for (int nh = 0; nh < 2; ++nh) {  // B from V^T: lane=N(h), K=j
                Frag vf;
                const unsigned char* vp =
                    smem + ALD_VT + (nh * 16 + l15) * 528 + jb * 64 + hiw * 32;
                vf.q[0] = *(const U4*)vp;
                vf.q[1] = *(const U4*)(vp + 16);
                oacc[mt][nh] = __builtin_amdgcn_wmma_f32_16x16x32_bf16(
                    false, pf.bf, false, vf.bf, (short)0, oacc[mt][nh], false, false);
            }
        }
    }
    // Epilogue: normalize, gate, scatter to Obuf [S][L][INNER] bf16.
    #pragma unroll
    for (int mt = 0; mt < 2; ++mt) {
        #pragma unroll
        for (int nh = 0; nh < 2; ++nh) {
            #pragma unroll
            for (int v = 0; v < 8; ++v) {
                int iloc = w * 32 + mt * 16 + v + hiw * 8;
                int h = nh * 16 + l15;
                float o = oacc[mt][nh][v] / lstat[mt][v];
                o *= bf2f(Gt[iloc * 32 + h]);
                obuf[((size_t)(iloc * 256 + c)) * 256 + n * 32 + h] = f2bf(o);
            }
        }
    }
}

// ---------------------------------------------------------------------------
// Kernel 5: output GEMM  out = Obuf @ Wo^T + b_o  (f32 out)
// ---------------------------------------------------------------------------
__global__ __launch_bounds__(256) void k_out(const unsigned short* __restrict__ obuf,
                                             const unsigned short* __restrict__ wot,
                                             const float* __restrict__ b_o,
                                             float* __restrict__ out) {
    __shared__ __align__(16) unsigned char smem[64 * 528];
    const int rb = blockIdx.x * 64;
    const int t = threadIdx.x, lane = t & 31, w = t >> 5;
    #pragma unroll
    for (int i = 0; i < 8; ++i) {
        int idx = t + i * 256;
        int row = idx >> 5, qq = idx & 31;
        U4 d = *(const U4*)(obuf + (size_t)(rb + row) * 256 + qq * 8);
        *(U4*)(smem + row * 528 + qq * 16) = d;
    }
    __syncthreads();
    const int l15 = lane & 15, hiw = lane >> 4;
    const int mt = w & 3, nh = w >> 2;
    v8f acc[8];
    v8f z = {0.f,0.f,0.f,0.f,0.f,0.f,0.f,0.f};
    #pragma unroll
    for (int i = 0; i < 8; ++i) acc[i] = z;
    #pragma unroll
    for (int ks = 0; ks < 8; ++ks) {
        Frag a;
        const unsigned char* ap = smem + (mt * 16 + l15) * 528 + ks * 64 + hiw * 16;
        a.q[0] = *(const U4*)ap;
        a.q[1] = *(const U4*)(ap + 32);
        #pragma unroll
        for (int nt = 0; nt < 8; ++nt) {
            int ncol = (nh * 8 + nt) * 16 + l15;
            const unsigned short* bp = wot + (size_t)ncol * 256 + ks * 32 + hiw * 8;
            Frag b;
            b.q[0] = *(const U4*)bp;
            b.q[1] = *(const U4*)(bp + 16);
            acc[nt] = __builtin_amdgcn_wmma_f32_16x16x32_bf16(
                false, a.bf, false, b.bf, (short)0, acc[nt], false, false);
        }
    }
    #pragma unroll
    for (int nt = 0; nt < 8; ++nt) {
        int j = (nh * 8 + nt) * 16 + l15;
        float bo = b_o[j];
        #pragma unroll
        for (int v = 0; v < 8; ++v) {
            int r = rb + mt * 16 + v + hiw * 8;
            out[(size_t)r * 256 + j] = acc[nt][v] + bo;
        }
    }
}

// ---------------------------------------------------------------------------
// Workspace layout (bytes):
//   [0, 32MB)          Xln bf16 (aliased later as Obuf bf16 — safe: stream order)
//   [32MB, +512KB)     Wt  bf16 [4][256 n][256 k]
//   [.., +128KB)       Wot bf16 [256 n][256 k]
//   [.., +128MB)       QKVG bf16 [4][L][H][S][Dh]
// Total ~168.4 MB.
// ---------------------------------------------------------------------------
extern "C" void kernel_launch(void* const* d_in, const int* in_sizes, int n_in,
                              void* d_out, int out_size, void* d_ws, size_t ws_size,
                              hipStream_t stream) {
    (void)in_sizes; (void)n_in; (void)out_size; (void)ws_size;
    const float* msa   = (const float*)d_in[0];
    const float* gamma = (const float*)d_in[1];
    const float* beta  = (const float*)d_in[2];
    const float* wq    = (const float*)d_in[3];
    const float* wk    = (const float*)d_in[4];
    const float* wv    = (const float*)d_in[5];
    const float* wg    = (const float*)d_in[6];
    const float* bg    = (const float*)d_in[7];
    const float* wo    = (const float*)d_in[8];
    const float* bo    = (const float*)d_in[9];

    char* ws = (char*)d_ws;
    unsigned short* xo   = (unsigned short*)(ws + 0);           // Xln / Obuf alias
    unsigned short* wt   = (unsigned short*)(ws + 33554432);
    unsigned short* wot  = (unsigned short*)(ws + 33554432 + 524288);
    unsigned short* qkvg = (unsigned short*)(ws + 33554432 + 524288 + 131072);

    k_ln   <<<8192, 256, 0, stream>>>(msa, gamma, beta, xo);
    k_wprep<<<dim3(64, 5), 256, 0, stream>>>(wq, wk, wv, wg, wo, wt, wot);
    k_proj <<<dim3(1024, 4), 256, 0, stream>>>(xo, wt, bg, qkvg);
    k_att  <<<2048, 256, 0, stream>>>(qkvg, xo);
    k_out  <<<1024, 256, 0, stream>>>(xo, wot, bo, (float*)d_out);
}
